// MHLA_42082089566579
// MI455X (gfx1250) — matrix operand order repaired and checked by reading, not compiled, not run on hardware
//
#include <hip/hip_runtime.h>
#include <hip/hip_bf16.h>
#include <stdint.h>

// ---------------------------------------------------------------------------
// MLA forward for MI455X (gfx1250, wave32, WMMA bf16 16x16x32, f32 accum)
// LDS-staged double-buffered GEMM with async global->LDS copies (ASYNCcnt).
// ---------------------------------------------------------------------------

typedef __bf16 bf16_t;
typedef __attribute__((ext_vector_type(16))) __bf16 v16bf;
typedef __attribute__((ext_vector_type(8)))  __bf16 v8bf;
typedef __attribute__((ext_vector_type(4)))  __bf16 v4bf;
typedef __attribute__((ext_vector_type(8)))  float  v8f;
typedef __attribute__((ext_vector_type(4)))  int    v4i;

#define BB 2
#define SEQ 4096
#define DM 2048
#define DL 512
#define TCACHE 4096
#define TTOT 8192

#if __has_builtin(__builtin_amdgcn_global_load_async_to_lds_b128)
#define HAVE_ASYNC_LDS 1
#else
#define HAVE_ASYNC_LDS 0
#endif

__device__ __forceinline__ v8f zero8() {
  v8f z = {0.f, 0.f, 0.f, 0.f, 0.f, 0.f, 0.f, 0.f};
  return z;
}

// Copy 16 bytes global -> LDS. Async (ASYNCcnt-tracked) when the toolchain
// exposes the gfx1250 builtin; synchronous fallback otherwise.
__device__ __forceinline__ void copy16_g2l(const bf16_t* __restrict__ g,
                                           bf16_t* __restrict__ l) {
#if HAVE_ASYNC_LDS
  __builtin_amdgcn_global_load_async_to_lds_b128((v4i*)g, (v4i*)l, 0, 0);
#else
  *(v8bf*)l = *(const v8bf*)g;
#endif
}

__device__ __forceinline__ void async_wait0() {
#if HAVE_ASYNC_LDS
#if __has_builtin(__builtin_amdgcn_s_wait_asynccnt)
  __builtin_amdgcn_s_wait_asynccnt(0);
#else
  asm volatile("s_wait_asynccnt 0x0" ::: "memory");
#endif
#endif
}

union AFragU { v16bf v; v8bf h[2]; };

// A fragment: 16x32 bf16, A row-major [M][lda].
// lane half h: VGPR0-3 = K[8h..8h+7], VGPR4-7 = K[16+8h..16+8h+7]; row = m0+(lane&15)
__device__ __forceinline__ v16bf load_a_frag(const bf16_t* __restrict__ A, int lda,
                                             int m0, int k0, int lane) {
  int h = lane >> 4, m = lane & 15;
  const bf16_t* p = A + (long)(m0 + m) * lda + k0 + 8 * h;
  AFragU f;
  f.h[0] = *(const v8bf*)p;
  f.h[1] = *(const v8bf*)(p + 16);
  return f.v;
}

// B fragment: 32x16 bf16, B supplied TRANSPOSED row-major Bt[N][ldb].
// lane half h: VGPR0-3 = K[8h..8h+7], VGPR4-7 = K[16+8h..16+8h+7]; col = n0+(lane&15)
__device__ __forceinline__ v16bf load_b_frag(const bf16_t* __restrict__ Bt, int ldb,
                                             int n0, int k0, int lane) {
  int h = lane >> 4, n = lane & 15;
  AFragU f;
  const bf16_t* p = Bt + (long)(n0 + n) * ldb + k0 + 8 * h;
  f.h[0] = *(const v8bf*)p;
  f.h[1] = *(const v8bf*)(p + 16);
  return f.v;
}

__device__ __forceinline__ v8f wmma_bf16(v16bf a, v16bf b, v8f c) {
  return __builtin_amdgcn_wmma_f32_16x16x32_bf16(false, a, false, b, (short)0, c,
                                                 false, false);
}

// ---------------------------------------------------------------------------
// Elementwise convert fp32 -> bf16 (4-wide)
// ---------------------------------------------------------------------------
__global__ void cvt_bf16_v4(const float* __restrict__ in, bf16_t* __restrict__ out,
                            long n) {
  long i = (((long)blockIdx.x * blockDim.x) + threadIdx.x) * 4;
  if (i + 3 >= n) return;
  float4 f = *(const float4*)(in + i);
  v4bf o = {(bf16_t)f.x, (bf16_t)f.y, (bf16_t)f.z, (bf16_t)f.w};
  *(v4bf*)(out + i) = o;
}

// convert + transpose: in[rows][cols] fp32 -> out[cols][rows] bf16
__global__ void cvt_transpose(const float* __restrict__ in, bf16_t* __restrict__ out,
                              int rows, int cols) {
  int c = blockIdx.x * 32 + (threadIdx.x & 31);
  int r = blockIdx.y * 8 + (threadIdx.x >> 5);
  if (r < rows && c < cols)
    out[(long)c * rows + r] = (bf16_t)in[(long)r * cols + c];
}

// cache -> l_kv fp32 output region, bf16 copy, bf16 transposed copy
__global__ void cache_expand(const float* __restrict__ cache,
                             float* __restrict__ out_lkv,
                             bf16_t* __restrict__ lkv_b,
                             bf16_t* __restrict__ lkv_t) {
  long b = blockIdx.z;
  int l = blockIdx.x * 32 + (threadIdx.x & 31);
  int t = blockIdx.y * 8 + (threadIdx.x >> 5);
  float v = cache[(b * TCACHE + t) * DL + l];
  out_lkv[(b * TTOT + t) * DL + l] = v;
  lkv_b[(b * TTOT + t) * DL + l] = (bf16_t)v;
  lkv_t[(b * DL + l) * TTOT + t] = (bf16_t)v;
}

// ---------------------------------------------------------------------------
// bf16 WMMA GEMM: C[M][N] = A[M][K] * Bt[N][K]^T
// block = 256 threads (8 waves); block tile 64(M) x 256(N), k-step 32.
// Double-buffered LDS staging (A: 4KB, B: 16KB per buffer = 40KB total),
// async global->LDS copies overlapped with WMMA on the other buffer.
// Wave tile: 32(M) x 64(N) -> 8 WMMA per k-step from 6 LDS fragments.
// ---------------------------------------------------------------------------
#define GM 64
#define GN 256
#define GK 32

__global__ void gemm_bf16(const bf16_t* __restrict__ A, long strideA, int lda,
                          const bf16_t* __restrict__ Bt, long strideB, int ldb,
                          float* __restrict__ Cf, long strideCf,
                          bf16_t* __restrict__ Cb, long strideCb, int ldc,
                          bf16_t* __restrict__ Cbt, long strideCbt, int ldt,
                          int K) {
  __shared__ __align__(16) bf16_t Ash[2][GM][GK];
  __shared__ __align__(16) bf16_t Bsh[2][GN][GK];

  int tid = threadIdx.x;
  int lane = tid & 31;
  int w = tid >> 5;
  long zb = blockIdx.z;
  const bf16_t* Ab = A + zb * strideA + (long)(blockIdx.x * GM) * lda;
  const bf16_t* Bb = Bt + zb * strideB + (long)(blockIdx.y * GN) * ldb;

  // copy assignments: 16B chunks; A tile = 256 chunks (1/thread),
  // B tile = 1024 chunks (4/thread)
  const int crow = tid >> 2;            // 0..63
  const int cchk = (tid & 3) * 8;       // element offset of 16B chunk

  v8f acc[2][4];
#pragma unroll
  for (int im = 0; im < 2; ++im)
#pragma unroll
    for (int j = 0; j < 4; ++j) acc[im][j] = zero8();

  // prologue: stage k=0 into buffer 0
  {
    copy16_g2l(Ab + (long)crow * lda + cchk, &Ash[0][crow][cchk]);
#pragma unroll
    for (int i = 0; i < 4; ++i)
      copy16_g2l(Bb + (long)(crow + 64 * i) * ldb + cchk,
                 &Bsh[0][crow + 64 * i][cchk]);
  }
  async_wait0();
  __syncthreads();

  const int nk = K / GK;
  const int ma = (w & 1) * 32;
  const int nb = (w >> 1) * 64;

  for (int kb = 0; kb < nk; ++kb) {
    int buf = kb & 1;
    if (kb + 1 < nk) {
      int k = (kb + 1) * GK;
      copy16_g2l(Ab + (long)crow * lda + k + cchk, &Ash[buf ^ 1][crow][cchk]);
#pragma unroll
      for (int i = 0; i < 4; ++i)
        copy16_g2l(Bb + (long)(crow + 64 * i) * ldb + k + cchk,
                   &Bsh[buf ^ 1][crow + 64 * i][cchk]);
    }
    const bf16_t* Abase = &Ash[buf][0][0];
    const bf16_t* Bbase = &Bsh[buf][0][0];
    v16bf a0 = load_a_frag(Abase, GK, ma, 0, lane);
    v16bf a1 = load_a_frag(Abase, GK, ma + 16, 0, lane);
#pragma unroll
    for (int j = 0; j < 4; ++j) {
      v16bf bfrag = load_b_frag(Bbase, GK, nb + 16 * j, 0, lane);
      acc[0][j] = wmma_bf16(a0, bfrag, acc[0][j]);
      acc[1][j] = wmma_bf16(a1, bfrag, acc[1][j]);
    }
    async_wait0();
    __syncthreads();
  }

  int h = lane >> 4, n = lane & 15;
#pragma unroll
  for (int im = 0; im < 2; ++im)
#pragma unroll
    for (int j = 0; j < 4; ++j)
#pragma unroll
      for (int r = 0; r < 8; ++r) {
        int m = blockIdx.x * GM + ma + im * 16 + r + 8 * h;
        int c = blockIdx.y * GN + nb + 16 * j + n;
        float v = acc[im][j][r];
        if (Cf)  Cf[zb * strideCf + (long)m * ldc + c] = v;
        if (Cb)  Cb[zb * strideCb + (long)m * ldc + c] = (bf16_t)v;
        if (Cbt) Cbt[zb * strideCbt + (long)c * ldt + m] = (bf16_t)v;
      }
}

// ---------------------------------------------------------------------------
// Flash attention in latent space.
// block = 256 threads (8 waves), one 16-query tile per block.
// lq tile staged to LDS once; key tile TK=128: score phase wave w -> key cols
// [16w,16w+16); ctx phase wave w -> latent cols [64w,64w+64) (VGPR-resident).
// ---------------------------------------------------------------------------
#define TK 128

__global__ void flash_latent(const bf16_t* __restrict__ lq,
                             const bf16_t* __restrict__ lkv_b,
                             const bf16_t* __restrict__ lkv_t,
                             bf16_t* __restrict__ ctx_b) {
  __shared__ __align__(16) bf16_t Qsh[16][DL];   // 16KB, staged once
  __shared__ __align__(16) float Ssh[16][TK];    // 8KB
  __shared__ __align__(16) bf16_t Psh[16][TK];   // 4KB
  __shared__ float red[16][16];
  __shared__ float mrow[16], lrow[16], arow[16];

  long b = blockIdx.y;
  int s0 = blockIdx.x * 16;
  int tid = threadIdx.x;
  int lane = tid & 31, w = tid >> 5;
  int row = tid >> 4, sub = tid & 15;

  const bf16_t* lqB = lq + b * (long)SEQ * DL;
  const bf16_t* kvB = lkv_b + b * (long)TTOT * DL;
  const bf16_t* kvT = lkv_t + b * (long)DL * TTOT;

  if (tid < 16) { mrow[tid] = -1e30f; lrow[tid] = 0.f; }

  // stage the 16x512 lq tile into LDS (1024 16B chunks, 4 per thread)
  {
    int qrow = tid >> 4;
    int qc = (tid & 15) * 32;
#pragma unroll
    for (int c = 0; c < 4; ++c)
      copy16_g2l(lqB + (long)(s0 + qrow) * DL + qc + 8 * c, &Qsh[qrow][qc + 8 * c]);
  }
  v8f acc[4];
#pragma unroll
  for (int j = 0; j < 4; ++j) acc[j] = zero8();
  async_wait0();
  __syncthreads();

  const float scale = 0.04419417382415922f;  // 1/sqrt(512)

  for (int t0 = 0; t0 < TTOT; t0 += TK) {
    // ---- scores subtile: 16 x 16 per wave, K = DL
    v8f sc = zero8();
    for (int k = 0; k < DL; k += 32) {
      v16bf a = load_a_frag(&Qsh[0][0], DL, 0, k, lane);
      v16bf bfrag = load_b_frag(kvB, DL, t0 + 16 * w, k, lane);
      sc = wmma_bf16(a, bfrag, sc);
    }
    {
      int h = lane >> 4, n = lane & 15;
#pragma unroll
      for (int r = 0; r < 8; ++r) Ssh[r + 8 * h][16 * w + n] = sc[r] * scale;
    }
    __syncthreads();

    // ---- online softmax over 16 x TK tile
    float lm = -1e30f;
#pragma unroll
    for (int c = 0; c < 8; ++c) lm = fmaxf(lm, Ssh[row][sub * 8 + c]);
    red[row][sub] = lm;
    __syncthreads();
    if (sub == 0) {
      float mx = mrow[row];
#pragma unroll
      for (int i = 0; i < 16; ++i) mx = fmaxf(mx, red[row][i]);
      arow[row] = __expf(mrow[row] - mx);
      mrow[row] = mx;
    }
    __syncthreads();
    {
      float mx = mrow[row];
      float ls = 0.f;
#pragma unroll
      for (int c = 0; c < 8; ++c) {
        float p = __expf(Ssh[row][sub * 8 + c] - mx);
        Psh[row][sub * 8 + c] = (bf16_t)p;
        ls += p;
      }
      red[row][sub] = ls;
    }
    __syncthreads();
    if (sub == 0) {
      float s = 0.f;
#pragma unroll
      for (int i = 0; i < 16; ++i) s += red[row][i];
      lrow[row] = arow[row] * lrow[row] + s;
    }
    __syncthreads();

    // ---- ctx accumulate: rescale then acc += P(16xTK) @ V(TK x 64-slice)
    {
      int h = lane >> 4;
      float al[8];
#pragma unroll
      for (int r = 0; r < 8; ++r) al[r] = arow[r + 8 * h];
#pragma unroll
      for (int j = 0; j < 4; ++j)
#pragma unroll
        for (int r = 0; r < 8; ++r) acc[j][r] *= al[r];

      for (int k = 0; k < TK; k += 32) {
        v16bf a = load_a_frag(&Psh[0][0], TK, 0, k, lane);
#pragma unroll
        for (int j = 0; j < 4; ++j) {
          v16bf bfrag = load_b_frag(kvT, TTOT, 64 * w + 16 * j, t0 + k, lane);
          acc[j] = wmma_bf16(a, bfrag, acc[j]);
        }
      }
    }
    __syncthreads();
  }

  // ---- finalize: divide by softmax denominator, emit bf16 ctx
  {
    int h = lane >> 4, n = lane & 15;
    float il[8];
#pragma unroll
    for (int r = 0; r < 8; ++r) il[r] = 1.f / lrow[r + 8 * h];
#pragma unroll
    for (int j = 0; j < 4; ++j)
#pragma unroll
      for (int r = 0; r < 8; ++r) {
        int m = s0 + r + 8 * h;
        int c = 64 * w + 16 * j + n;
        ctx_b[(b * SEQ + m) * (long)DL + c] = (bf16_t)(acc[j][r] * il[r]);
      }
  }
}

// ---------------------------------------------------------------------------
// Host-side launcher
// ---------------------------------------------------------------------------
extern "C" void kernel_launch(void* const* d_in, const int* in_sizes, int n_in,
                              void* d_out, int out_size, void* d_ws, size_t ws_size,
                              hipStream_t stream) {
  const float* x     = (const float*)d_in[0];
  const float* cache = (const float*)d_in[1];
  const float* W_kv  = (const float*)d_in[2];
  const float* W_q   = (const float*)d_in[3];
  const float* W_lq  = (const float*)d_in[4];
  const float* W_o   = (const float*)d_in[5];

  float* out = (float*)d_out;                          // [B][S][DM]
  float* out_lkv = out + (long)BB * SEQ * DM;          // [B][TTOT][DL]

  // workspace layout (bytes)
  char* ws = (char*)d_ws;
  bf16_t* xb      = (bf16_t*)(ws + 0);          //  B*S*DM        33.6 MB
  bf16_t* Wkv_t   = (bf16_t*)(ws + 33554432);   //  [DL][DM]       2 MB
  bf16_t* Wq_b    = (bf16_t*)(ws + 35651584);   //  [DM][DM]       8 MB
  bf16_t* Wlq_t   = (bf16_t*)(ws + 44040192);   //  [DL][DM]       2 MB
  bf16_t* Wo_t    = (bf16_t*)(ws + 46137344);   //  [DM][DL]       2 MB
  bf16_t* Wcomb_t = (bf16_t*)(ws + 48234496);   //  [DL][DM]       2 MB
  bf16_t* lq_b    = (bf16_t*)(ws + 50331648);   //  [B][S][DL]     8 MB
  bf16_t* lkv_b   = (bf16_t*)(ws + 58720256);   //  [B][TTOT][DL] 16 MB
  bf16_t* lkv_t   = (bf16_t*)(ws + 75497472);   //  [B][DL][TTOT] 16 MB
  bf16_t* ctx_b   = (bf16_t*)(ws + 92274688);   //  [B][S][DL]     8 MB
  (void)ws_size; (void)in_sizes; (void)n_in; (void)out_size;

  // 1) converts / transposes
  {
    long n = (long)BB * SEQ * DM;
    cvt_bf16_v4<<<dim3((n / 4 + 255) / 256), 256, 0, stream>>>(x, xb, n);
    long nq = (long)DM * DM;
    cvt_bf16_v4<<<dim3((nq / 4 + 255) / 256), 256, 0, stream>>>(W_q, Wq_b, nq);
    cvt_transpose<<<dim3(DL / 32, DM / 8), 256, 0, stream>>>(W_kv, Wkv_t, DM, DL);
    cvt_transpose<<<dim3(DL / 32, DM / 8), 256, 0, stream>>>(W_lq, Wlq_t, DM, DL);
    cvt_transpose<<<dim3(DM / 32, DL / 8), 256, 0, stream>>>(W_o, Wo_t, DL, DM);
    cache_expand<<<dim3(DL / 32, TCACHE / 8, BB), 256, 0, stream>>>(
        cache, out_lkv, lkv_b, lkv_t);
  }

  // 2) Wcomb_t[DL][DM] = (W_q @ W_lq)^T = W_lq^T @ W_q^T  (q/lq fusion)
  gemm_bf16<<<dim3(DL / GM, DM / GN, 1), 256, 0, stream>>>(
      Wlq_t, 0, DM, Wq_b, 0, DM,
      nullptr, 0, Wcomb_t, 0, DM, nullptr, 0, 0, DM);

  // 3) l_kv_new = x @ W_kv  (fp32 -> output, bf16 + bf16^T -> workspace)
  gemm_bf16<<<dim3(SEQ / GM, DL / GN, BB), 256, 0, stream>>>(
      xb, (long)SEQ * DM, DM, Wkv_t, 0, DM,
      out_lkv + (long)TCACHE * DL, (long)TTOT * DL,
      lkv_b + (long)TCACHE * DL, (long)TTOT * DL, DL,
      lkv_t + TCACHE, (long)DL * TTOT, TTOT, DM);

  // 4) lq = x @ Wcomb
  gemm_bf16<<<dim3(SEQ / GM, DL / GN, BB), 256, 0, stream>>>(
      xb, (long)SEQ * DM, DM, Wcomb_t, 0, DM,
      nullptr, 0, lq_b, (long)SEQ * DL, DL, nullptr, 0, 0, DM);

  // 5) flash attention in latent space -> ctx (bf16)
  flash_latent<<<dim3(SEQ / 16, BB), 256, 0, stream>>>(lq_b, lkv_b, lkv_t, ctx_b);

  // 6) out = ctx @ W_o (fp32 -> output)
  gemm_bf16<<<dim3(SEQ / GM, DM / GN, BB), 256, 0, stream>>>(
      ctx_b, (long)SEQ * DL, DL, Wo_t, 0, DL,
      out, (long)SEQ * DM, nullptr, 0, DM, nullptr, 0, 0, DL);
}
